// MRIEncoder_28226525069456
// MI455X (gfx1250) — matrix operand order repaired and checked by reading, compiled, not run on hardware
//
#include <hip/hip_runtime.h>

#define NODES 113
#define GRAPHS 4096

typedef __attribute__((ext_vector_type(16))) __bf16 v16bf;
typedef __attribute__((ext_vector_type(8)))  float  v8f;

union ABFrag { unsigned int u[8]; v16bf v; };

__device__ __forceinline__ unsigned short f2bf(float f){
  unsigned int u = __float_as_uint(f);
  u += 0x7FFFu + ((u >> 16) & 1u);           // round-to-nearest-even
  return (unsigned short)(u >> 16);
}
__device__ __forceinline__ unsigned int pack2(float a, float b){
  return (unsigned int)f2bf(a) | ((unsigned int)f2bf(b) << 16);
}
// monotonic float<->uint mapping so max() can use integer ds atomics (deterministic)
__device__ __forceinline__ unsigned int fkey(float f){
  unsigned int u = __float_as_uint(f);
  return (u & 0x80000000u) ? ~u : (u | 0x80000000u);
}
__device__ __forceinline__ float keyf(unsigned int k){
  unsigned int u = (k & 0x80000000u) ? (k & 0x7FFFFFFFu) : ~k;
  return __uint_as_float(u);
}

// A-matrix 16x32 bf16 fragment (ISA 7.12.2): lane&15 = M row; lanes 16-31 K+8;
// VGPR v: K = ((v>>2)<<4) + ((v&3)<<1). LDS stored as packed uint pairs.
__device__ __forceinline__ void loadA(const unsigned int* src, int rowBase, int kbHalf,
                                      int strideU, unsigned int* u){
  int lane = threadIdx.x & 31;
  const unsigned int* base = src + (rowBase + (lane & 15)) * strideU + kbHalf + ((lane >> 4) << 2);
  #pragma unroll
  for (int v = 0; v < 8; ++v) u[v] = base[((v >> 2) << 3) + (v & 3)];
}
// B-matrix 32x16 bf16 fragment: lane&15 = N col; lanes 16-31 K+8; VGPR v: K = ((v>>2)<<4) + ((v&3)<<1).
// Weights pre-transposed to [N][K] bf16, read as packed uints.
__device__ __forceinline__ void loadB(const unsigned int* src, int colBase, int kbHalf,
                                      int strideU, unsigned int* u){
  int lane = threadIdx.x & 31;
  const unsigned int* base = src + (colBase + (lane & 15)) * strideU + kbHalf + ((lane >> 4) << 2);
  #pragma unroll
  for (int v = 0; v < 8; ++v) u[v] = base[((v >> 2) << 3) + (v & 3)];
}

// ---- K0: transpose + bf16-convert weights: Wt[n][k] = bf16(W[k][n]) ----
__global__ __launch_bounds__(256) void k0_pack(const float* __restrict__ W2, const float* __restrict__ W3,
                                               const float* __restrict__ W4,
                                               unsigned short* __restrict__ w2t,
                                               unsigned short* __restrict__ w3t,
                                               unsigned short* __restrict__ w4t){
  int i = blockIdx.x * 256 + threadIdx.x;
  if (i < 16384){            // W2 [128][128]
    int n = i >> 7, k = i & 127;
    w2t[n*128 + k] = f2bf(W2[k*128 + n]);
  } else if (i < 81920){     // W3 [128][512]
    int j = i - 16384; int n = j >> 7, k = j & 127;
    w3t[n*128 + k] = f2bf(W3[k*512 + n]);
  } else {                   // W4 [512][1024]
    int j = i - 81920; int n = j >> 9, k = j & 511;
    w4t[n*512 + k] = f2bf(W4[k*1024 + n]);
  }
}

// ---- K1: per-graph fused node-net + segment-max pool (1 block / graph) ----
__global__ __launch_bounds__(256) void k1_node(const float* __restrict__ x, const float* __restrict__ roi,
    const float* __restrict__ W1, const float* __restrict__ b1,
    const float* __restrict__ gng, const float* __restrict__ gnb,
    const unsigned int* __restrict__ w2t, const float* __restrict__ b2,
    unsigned int* __restrict__ zbf){
  __shared__ float xs[340];                 // 113 nodes x 3, pre-scaled
  __shared__ unsigned int hbf[128 * 66];    // 128x128 bf16 acts, padded stride (bank-conflict free)
  __shared__ unsigned int zkey[128];
  int g = blockIdx.x, tid = threadIdx.x;
  if (tid < 339) xs[tid] = x[g*339 + tid] * roi[tid];
  if (tid < 128) zkey[tid] = 0u;            // key 0 == -NaN: identity for max
  __syncthreads();
  // Linear(3->128) + GroupNorm(8 groups of 16) + LeakyReLU(0.2), direct to bf16
  for (int uix = tid; uix < 1024; uix += 256){
    int r = uix >> 3, gi = uix & 7, c0 = gi << 4;
    if (r < NODES){
      float h[16], s = 0.f, ss = 0.f;
      float x0 = xs[r*3], x1 = xs[r*3+1], x2 = xs[r*3+2];
      #pragma unroll
      for (int j = 0; j < 16; ++j){
        int c = c0 + j;
        float v = b1[c] + x0*W1[c] + x1*W1[128+c] + x2*W1[256+c];
        h[j] = v; s += v; ss += v*v;
      }
      float mu = s * 0.0625f;
      float inv = rsqrtf(ss * 0.0625f - mu*mu + 1e-5f);
      #pragma unroll
      for (int j = 0; j < 16; j += 2){
        int c = c0 + j;
        float a = (h[j]  -mu)*inv*gng[c]   + gnb[c];   a = a > 0.f ? a : 0.2f*a;
        float b = (h[j+1]-mu)*inv*gng[c+1] + gnb[c+1]; b = b > 0.f ? b : 0.2f*b;
        hbf[r*66 + (c >> 1)] = pack2(a, b);
      }
    } else {
      #pragma unroll
      for (int j = 0; j < 16; j += 2) hbf[r*66 + ((c0 + j) >> 1)] = 0u;
    }
  }
  __syncthreads();
  // GEMM2: 128x128 @ 128x128 bf16 WMMA; wave w owns M-tile w
  int lane = tid & 31, w = tid >> 5;
  int mrow = w << 4;
  ABFrag A[4];
  #pragma unroll
  for (int kt = 0; kt < 4; ++kt) loadA(hbf, mrow, kt*16, 66, A[kt].u);
  int mb = (lane >> 4) << 3;
  const float NEG_INF = __uint_as_float(0xff800000u);
  for (int nt = 0; nt < 8; ++nt){
    v8f acc = {};
    #pragma unroll
    for (int kt = 0; kt < 4; ++kt){
      ABFrag B;
      loadB(w2t, nt << 4, kt*16, 64, B.u);
      acc = __builtin_amdgcn_wmma_f32_16x16x32_bf16(false, A[kt].v, false, B.v,
                                                    (short)0, acc, false, false);
    }
    // branch-free in-register max over this wave's 8 rows (invalid rows -> -inf)
    float cmax = NEG_INF;
    #pragma unroll
    for (int v = 0; v < 8; ++v){
      float val = (mrow + mb + v < NODES) ? acc[v] : NEG_INF;   // v_cndmask, no EXEC games
      cmax = fmaxf(cmax, val);
    }
    // combine lane L with lane L^16 (same N column, other 8 rows)
    cmax = fmaxf(cmax, __shfl_xor(cmax, 16, 32));
    int n = (nt << 4) + (lane & 15);
    // all 32 lanes write the same value per column: idempotent max, deterministic
    atomicMax(&zkey[n], fkey(cmax + b2[n]));                    // one ds_max_u32 per lane
  }
  __syncthreads();
  if (tid < 64) zbf[g*64 + tid] = pack2(keyf(zkey[tid*2]), keyf(zkey[tid*2 + 1]));
}

// ---- K2: p = z @ W3 + b3  (4096x128x512, WMMA) ----
__global__ __launch_bounds__(256) void k2_proj(const unsigned int* __restrict__ zbf,
    const unsigned int* __restrict__ w3t, const float* __restrict__ b3,
    float* __restrict__ p){
  __shared__ unsigned int zb[128 * 66];
  int g0 = blockIdx.x << 7, tid = threadIdx.x;
  for (int e = tid; e < 128*64; e += 256){
    int r = e >> 6, cu = e & 63;
    zb[r*66 + cu] = zbf[(g0 + r)*64 + cu];
  }
  __syncthreads();
  int lane = tid & 31, w = tid >> 5;
  int mb = (lane >> 4) << 3;
  for (int t = w; t < 256; t += 8){
    int mt = t >> 5, nt = t & 31;
    v8f acc = {};
    #pragma unroll
    for (int kt = 0; kt < 4; ++kt){
      ABFrag A, B;
      loadA(zb, mt << 4, kt*16, 66, A.u);
      loadB(w3t, nt << 4, kt*16, 64, B.u);
      acc = __builtin_amdgcn_wmma_f32_16x16x32_bf16(false, A.v, false, B.v,
                                                    (short)0, acc, false, false);
    }
    int n = (nt << 4) + (lane & 15);
    float bb = b3[n];
    int mrowg = g0 + (mt << 4) + mb;
    #pragma unroll
    for (int v = 0; v < 8; ++v) p[(mrowg + v)*512 + n] = acc[v] + bb;
  }
}

// ---- K2b: BatchNorm batch stats (deterministic tree reduce), fused to scale/shift ----
__global__ __launch_bounds__(256) void k2b_stats(const float* __restrict__ p,
    const float* __restrict__ bng, const float* __restrict__ bnb,
    float* __restrict__ bnscale, float* __restrict__ bnshift){
  __shared__ float rs[256], rss[256];
  int c = blockIdx.x, tid = threadIdx.x;
  float s = 0.f, ss = 0.f;
  for (int r = tid; r < 4096; r += 256){ float v = p[r*512 + c]; s += v; ss += v*v; }
  rs[tid] = s; rss[tid] = ss;
  __syncthreads();
  for (int o = 128; o > 0; o >>= 1){
    if (tid < o){ rs[tid] += rs[tid + o]; rss[tid] += rss[tid + o]; }
    __syncthreads();
  }
  if (tid == 0){
    float mu  = rs[0]  * (1.f/4096.f);
    float var = rss[0] * (1.f/4096.f) - mu*mu;
    float sc  = bng[c] * rsqrtf(var + 1e-5f);
    bnscale[c] = sc;
    bnshift[c] = bnb[c] - mu*sc;
  }
}

// ---- K3: out = relu(BN(p)) @ W4 + b4, then row L2-normalize (32 rows / block) ----
__global__ __launch_bounds__(256) void k3_out(const float* __restrict__ p,
    const float* __restrict__ bnscale, const float* __restrict__ bnshift,
    const unsigned int* __restrict__ w4t, const float* __restrict__ b4,
    float* __restrict__ out){
  __shared__ unsigned int pnb[32 * 258];    // 32x512 bf16, padded stride
  __shared__ float part[32][8];
  __shared__ float rinv[32];
  int g0 = blockIdx.x << 5, tid = threadIdx.x;
  for (int e = tid; e < 32*256; e += 256){
    int r = e >> 8, cp = e & 255, c = cp << 1;
    int base = (g0 + r)*512 + c;
    float a = fmaxf(p[base]     * bnscale[c]     + bnshift[c],     0.f);
    float b = fmaxf(p[base + 1] * bnscale[c + 1] + bnshift[c + 1], 0.f);
    pnb[r*258 + cp] = pack2(a, b);
  }
  __syncthreads();
  int lane = tid & 31, w = tid >> 5;
  int mt = w & 1;
  int mb = (mt << 4) + ((lane >> 4) << 3);
  for (int t = w; t < 128; t += 8){
    int nt = t >> 1;
    v8f acc = {};
    for (int kt = 0; kt < 16; ++kt){
      ABFrag A, B;
      loadA(pnb, mt << 4, kt*16, 258, A.u);
      loadB(w4t, nt << 4, kt*16, 256, B.u);
      acc = __builtin_amdgcn_wmma_f32_16x16x32_bf16(false, A.v, false, B.v,
                                                    (short)0, acc, false, false);
    }
    int n = (nt << 4) + (lane & 15);
    float bb = b4[n];
    #pragma unroll
    for (int v = 0; v < 8; ++v) out[(g0 + mb + v)*1024 + n] = acc[v] + bb;
  }
  __threadfence();
  __syncthreads();
  { // deterministic per-row sum of squares: 8 fixed segments per row, fixed combine order
    int r = tid >> 3, seg = tid & 7;
    float s = 0.f;
    const float* row = out + (g0 + r)*1024 + (seg << 7);
    for (int c = 0; c < 128; ++c){ float v = row[c]; s += v*v; }
    part[r][seg] = s;
  }
  __syncthreads();
  if (tid < 32){
    float s = 0.f;
    #pragma unroll
    for (int j = 0; j < 8; ++j) s += part[tid][j];
    rinv[tid] = 1.f / fmaxf(sqrtf(s), 1e-12f);
  }
  __syncthreads();
  for (int e = tid; e < 32*1024; e += 256){
    int r = e >> 10;
    int idx = (g0 + r)*1024 + (e & 1023);
    out[idx] *= rinv[r];
  }
}

extern "C" void kernel_launch(void* const* d_in, const int* in_sizes, int n_in,
                              void* d_out, int out_size, void* d_ws, size_t ws_size,
                              hipStream_t stream){
  const float* x   = (const float*)d_in[0];
  const float* roi = (const float*)d_in[1];
  const float* W1  = (const float*)d_in[2];
  const float* b1  = (const float*)d_in[3];
  const float* gng = (const float*)d_in[4];
  const float* gnb = (const float*)d_in[5];
  const float* W2  = (const float*)d_in[6];
  const float* b2  = (const float*)d_in[7];
  const float* W3  = (const float*)d_in[8];
  const float* b3  = (const float*)d_in[9];
  const float* bng = (const float*)d_in[10];
  const float* bnb = (const float*)d_in[11];
  const float* W4  = (const float*)d_in[12];
  const float* b4  = (const float*)d_in[13];
  // d_in[14] = batch ids: unused, nodes are graph-major contiguous (113 per graph).

  char* ws = (char*)d_ws;
  unsigned short* w2t = (unsigned short*)(ws);              //  32 KB
  unsigned short* w3t = (unsigned short*)(ws +    32768);   // 128 KB
  unsigned short* w4t = (unsigned short*)(ws +   163840);   //   1 MB
  unsigned int*   zbf = (unsigned int*)  (ws +  1212416);   //   1 MB (4096x128 bf16)
  float*          pp  = (float*)         (ws +  2260992);   //   8 MB (4096x512 f32)
  float*      bnscale = (float*)         (ws + 10649600);
  float*      bnshift = (float*)         (ws + 10651648);

  k0_pack <<<2368,  256, 0, stream>>>(W2, W3, W4, w2t, w3t, w4t);
  k1_node <<<GRAPHS,256, 0, stream>>>(x, roi, W1, b1, gng, gnb,
                                      (const unsigned int*)w2t, b2, zbf);
  k2_proj <<<32,    256, 0, stream>>>(zbf, (const unsigned int*)w3t, b3, pp);
  k2b_stats<<<512,  256, 0, stream>>>(pp, bng, bnb, bnscale, bnshift);
  k3_out  <<<128,   256, 0, stream>>>(pp, bnscale, bnshift,
                                      (const unsigned int*)w4t, b4, (float*)d_out);
}
